// BaseGCN_46042049413263
// MI455X (gfx1250) — matrix operand order repaired and checked by reading, not compiled, not run on hardware
//
#include <hip/hip_runtime.h>
#include <hip/hip_bf16.h>
#include <stdint.h>

#define NNODES   100000
#define NEDGES   3200000
#define NGRAPHS  256
#define KIN      128
#define HID      64
#define BN_EPS_F 1e-5f
#define L2_EPS_F 1e-12f

typedef __attribute__((ext_vector_type(16))) _Float16 v16h;
typedef __attribute__((ext_vector_type(8)))  float    v8f;

__device__ __forceinline__ float wave_sum32(float v) {
#pragma unroll
  for (int off = 16; off > 0; off >>= 1) v += __shfl_xor(v, off, 32);
  return v;
}

__device__ __forceinline__ void atomAddF(float* p, float v) {
  __hip_atomic_fetch_add(p, v, __ATOMIC_RELAXED, __HIP_MEMORY_SCOPE_AGENT);
}

// ---------------- degree / normalization ----------------
__global__ void k_deg_init(float* __restrict__ deg) {
  int i = blockIdx.x * blockDim.x + threadIdx.x;
  if (i < NNODES) deg[i] = 1.0f;  // self loop contribution
}

__global__ void k_deg_count(const long long* __restrict__ dst, float* __restrict__ deg) {
  int e = blockIdx.x * blockDim.x + threadIdx.x;
  if (e < NEDGES) atomAddF(&deg[(int)dst[e]], 1.0f);
}

__global__ void k_dinv(float* __restrict__ deg) {
  int i = blockIdx.x * blockDim.x + threadIdx.x;
  if (i < NNODES) {
    float d = deg[i];
    deg[i] = (d > 0.0f) ? rsqrtf(d) : 0.0f;  // in place: deg -> dinv
  }
}

// ---------------- weight pre-pack into WMMA B-fragment layout ----------------
// B fragment (32x16 f16, ISA 7.12.2): lane l: col = nt*16 + (l&15),
// element j -> k = k0 + (l>>4)*8 + j (j<8), k0 + 16 + (l>>4)*8 + (j-8) (j>=8).  Packed address:
//   Wp[ ((ks*4 + nt)*32 + lane)*16 + j ]
__global__ void k_pack_w(const float* __restrict__ W, _Float16* __restrict__ Wp, int K) {
  int t = blockIdx.x * blockDim.x + threadIdx.x;
  if (t >= K * HID) return;
  const int j    = t & 15;
  const int lane = (t >> 4) & 31;
  const int nt   = (t >> 9) & 3;
  const int ks   = t >> 11;
  const int half = lane >> 4, l16 = lane & 15;
  const int k = (ks << 5) + (half << 3) + (j < 8 ? j : 8 + j);
  const int n = (nt << 4) + l16;
  Wp[t] = (_Float16)W[(size_t)k * HID + n];
}

// ---------------- WMMA GEMM: [NNODES x K] @ [K x 64] ----------------
// One wave = 16-row x 64-col tile, 4 f32 accumulators, K unrolled (K/32 steps).
// A fragment (16-bit A 16x32): lane l holds row M = m0 + (l&15); per K-step its
// elements are two contiguous 8-float spans: k0 + half*8 + [0..7] and
// k0 + 16 + half*8 + [0..7]  -> 4x float4 loads + 8x v_cvt_pk_f16_f32.
// B fragment: one aligned 32-byte vector load from the pre-packed fp16 buffer.
// C/D: VGPR r, lane l -> row = m0 + r + (l>=16 ? 8 : 0), col = nt*16 + (l&15).
// MODE: 0 = raw store, 1 = bias + BN(eval) + ReLU, 2 = bias only
template <int K, int MODE>
__global__ __launch_bounds__(256) void k_gemm_wmma(
    const float* __restrict__ A, const _Float16* __restrict__ Wp,
    const float* __restrict__ bias, const float* __restrict__ gamma,
    const float* __restrict__ beta, float* __restrict__ out) {
  const int lane = threadIdx.x & 31;
  const int wave = (int)((blockIdx.x * (unsigned)blockDim.x + threadIdx.x) >> 5);
  if (wave >= NNODES / 16) return;
  const int m0   = wave << 4;
  const int half = lane >> 4;
  const int l16  = lane & 15;

  v8f acc[4] = {};
  const float* arow = A + (size_t)(m0 + l16) * K + (half << 3);

#pragma unroll
  for (int ks = 0; ks < K / 32; ++ks) {
    const int k0 = ks << 5;
    const float4 a0 = *(const float4*)(arow + k0);
    const float4 a1 = *(const float4*)(arow + k0 + 4);
    const float4 a2 = *(const float4*)(arow + k0 + 16);
    const float4 a3 = *(const float4*)(arow + k0 + 20);
    v16h a;
    a[0]  = (_Float16)a0.x; a[1]  = (_Float16)a0.y;
    a[2]  = (_Float16)a0.z; a[3]  = (_Float16)a0.w;
    a[4]  = (_Float16)a1.x; a[5]  = (_Float16)a1.y;
    a[6]  = (_Float16)a1.z; a[7]  = (_Float16)a1.w;
    a[8]  = (_Float16)a2.x; a[9]  = (_Float16)a2.y;
    a[10] = (_Float16)a2.z; a[11] = (_Float16)a2.w;
    a[12] = (_Float16)a3.x; a[13] = (_Float16)a3.y;
    a[14] = (_Float16)a3.z; a[15] = (_Float16)a3.w;
#pragma unroll
    for (int nt = 0; nt < 4; ++nt) {
      const v16h b = *(const v16h*)(Wp + ((size_t)(((ks << 2) + nt) << 5) + lane) * 16);
      acc[nt] = __builtin_amdgcn_wmma_f32_16x16x32_f16(
          false, a, false, b, (short)0, acc[nt], false, false);
    }
  }

#pragma unroll
  for (int nt = 0; nt < 4; ++nt) {
    const int col = (nt << 4) + l16;
    float bv = 0.0f, gs = 1.0f, bb = 0.0f;
    if constexpr (MODE != 0) bv = bias[col];
    if constexpr (MODE == 1) { gs = gamma[col] * rsqrtf(1.0f + BN_EPS_F); bb = beta[col]; }
#pragma unroll
    for (int r = 0; r < 8; ++r) {
      const int row = m0 + (half << 3) + r;
      float v = acc[nt][r] + bv;
      if constexpr (MODE == 1) { v = v * gs + bb; v = (v > 0.0f) ? v : 0.0f; }
      out[(size_t)row * HID + col] = v;
    }
  }
}

// ---------------- aggregation ----------------
// out[i] = h[i] * dinv[i]^2   (self-loop term; fully overwrites accumulator)
__global__ void k_self_init(const float* __restrict__ h, const float* __restrict__ dinv,
                            float* __restrict__ out) {
  int t = blockIdx.x * blockDim.x + threadIdx.x;
  if (t >= NNODES * (HID / 4)) return;
  const int i = t / (HID / 4);
  const int c = (t % (HID / 4)) * 4;
  const float di = dinv[i];
  const float s = di * di;
  const float4 hv = *(const float4*)(h + (size_t)i * HID + c);
  float4 o;
  o.x = hv.x * s; o.y = hv.y * s; o.z = hv.z * s; o.w = hv.w * s;
  *(float4*)(out + (size_t)i * HID + c) = o;
}

// One wave per edge: coalesced 256B gather of h[src], 64 fp32 L2 atomics to out[dst].
__global__ void k_edge_agg(const long long* __restrict__ src, const long long* __restrict__ dst,
                           const float* __restrict__ h, const float* __restrict__ dinv,
                           float* __restrict__ out) {
  const long long tid = (long long)blockIdx.x * blockDim.x + threadIdx.x;
  const int e = (int)(tid >> 5);
  if (e >= NEDGES) return;
  const int lane = threadIdx.x & 31;
  const int s = (int)src[e];
  const int d = (int)dst[e];
  const float scale = dinv[s] * dinv[d];
  const float2 hv = *(const float2*)(h + (size_t)s * HID + lane * 2);
  float* op = out + (size_t)d * HID + lane * 2;
  atomAddF(op,     hv.x * scale);
  atomAddF(op + 1, hv.y * scale);
}

// x = relu((x + b) * (g/sqrt(1+eps)) + beta), in place
__global__ void k_post(float* __restrict__ x, const float* __restrict__ b,
                       const float* __restrict__ g, const float* __restrict__ be) {
  int t = blockIdx.x * blockDim.x + threadIdx.x;
  if (t >= NNODES * HID) return;
  const int c = t & (HID - 1);
  const float gs = g[c] * rsqrtf(1.0f + BN_EPS_F);
  float v = (x[t] + b[c]) * gs + be[c];
  x[t] = (v > 0.0f) ? v : 0.0f;
}

// ---------------- normalization / pooling ----------------
__global__ void k_l2rows(const float* __restrict__ in, float* __restrict__ out) {
  const int wave = (int)((blockIdx.x * (unsigned)blockDim.x + threadIdx.x) >> 5);
  if (wave >= NNODES) return;
  const int lane = threadIdx.x & 31;
  const float2 v = *(const float2*)(in + (size_t)wave * HID + lane * 2);
  const float ss = wave_sum32(v.x * v.x + v.y * v.y);
  const float inv = 1.0f / fmaxf(sqrtf(ss), L2_EPS_F);
  float2 o; o.x = v.x * inv; o.y = v.y * inv;
  *(float2*)(out + (size_t)wave * HID + lane * 2) = o;
}

__global__ void k_pool_zero(float* __restrict__ sums, float* __restrict__ cnts) {
  int t = blockIdx.x * blockDim.x + threadIdx.x;
  if (t < NGRAPHS * HID) sums[t] = 0.0f;
  if (t < NGRAPHS) cnts[t] = 0.0f;
}

__global__ void k_pool_acc(const float* __restrict__ emb, const long long* __restrict__ batch,
                           float* __restrict__ sums, float* __restrict__ cnts) {
  const int wave = (int)((blockIdx.x * (unsigned)blockDim.x + threadIdx.x) >> 5);
  if (wave >= NNODES) return;
  const int lane = threadIdx.x & 31;
  const int g = (int)batch[wave];
  const float2 v = *(const float2*)(emb + (size_t)wave * HID + lane * 2);
  float* sp = sums + (size_t)g * HID + lane * 2;
  atomAddF(sp, v.x);
  atomAddF(sp + 1, v.y);
  if (lane == 0) atomAddF(&cnts[g], 1.0f);
}

__global__ void k_graph_emb(const float* __restrict__ sums, const float* __restrict__ cnts,
                            float* __restrict__ out) {
  const int wave = (int)((blockIdx.x * (unsigned)blockDim.x + threadIdx.x) >> 5);
  if (wave >= NGRAPHS) return;
  const int lane = threadIdx.x & 31;
  const float cnt = fmaxf(cnts[wave], 1.0f);
  float2 v = *(const float2*)(sums + (size_t)wave * HID + lane * 2);
  v.x /= cnt; v.y /= cnt;
  const float ss = wave_sum32(v.x * v.x + v.y * v.y);
  const float inv = 1.0f / fmaxf(sqrtf(ss), L2_EPS_F);
  float2 o; o.x = v.x * inv; o.y = v.y * inv;
  *(float2*)(out + (size_t)wave * HID + lane * 2) = o;
}

// ---------------- launch ----------------
extern "C" void kernel_launch(void* const* d_in, const int* in_sizes, int n_in,
                              void* d_out, int out_size, void* d_ws, size_t ws_size,
                              hipStream_t stream) {
  (void)in_sizes; (void)n_in; (void)out_size; (void)ws_size;

  const float*     x     = (const float*)d_in[0];
  const long long* ei    = (const long long*)d_in[1];   // [2, E] int64
  const long long* batch = (const long long*)d_in[2];   // [N] int64
  const float* w1  = (const float*)d_in[3];
  const float* b1  = (const float*)d_in[4];
  const float* g1  = (const float*)d_in[5];
  const float* be1 = (const float*)d_in[6];
  const float* w2  = (const float*)d_in[7];
  const float* b2  = (const float*)d_in[8];
  const float* g2  = (const float*)d_in[9];
  const float* be2 = (const float*)d_in[10];
  const float* wp1 = (const float*)d_in[11];
  const float* bp1 = (const float*)d_in[12];
  const float* gp  = (const float*)d_in[13];
  const float* bep = (const float*)d_in[14];
  const float* wp2 = (const float*)d_in[15];
  const float* bp2 = (const float*)d_in[16];

  const long long* esrc = ei;
  const long long* edst = ei + NEDGES;

  float* out = (float*)d_out;               // [N*64] emb, then [256*64] graph emb
  float* emb = out;
  float* gemb = out + (size_t)NNODES * HID;

  // workspace carve-up (all fully rewritten every call -> replay safe)
  float* ws   = (float*)d_ws;
  float* dinv = ws;                                              // N (deg then dinv)
  float* bufA = ws + (((size_t)NNODES + 255) / 256) * 256;       // N*64
  float* bufB = bufA + (size_t)NNODES * HID;                     // N*64
  float* sums = bufB + (size_t)NNODES * HID;                     // 256*64
  float* cnts = sums + (size_t)NGRAPHS * HID;                    // 256
  _Float16* w1p  = (_Float16*)(cnts + NGRAPHS);                  // 128*64 halves (32B aligned)
  _Float16* w2p  = w1p + KIN * HID;                              // 64*64
  _Float16* wp1p = w2p + HID * HID;
  _Float16* wp2p = wp1p + HID * HID;

  const int TB = 256;
  const int blkN    = (NNODES + TB - 1) / TB;
  const int blkE    = (NEDGES + TB - 1) / TB;
  const int blkGemm = ((NNODES / 16) * 32 + TB - 1) / TB;        // wave per 16-row tile
  const int blkSelf = (NNODES * (HID / 4) + TB - 1) / TB;
  const int blkEdge = (int)(((long long)NEDGES * 32 + TB - 1) / TB);
  const int blkElem = (NNODES * HID + TB - 1) / TB;
  const int blkRowW = (NNODES * 32 + TB - 1) / TB;               // wave per row
  const int blkPool = (NGRAPHS * HID + TB - 1) / TB;
  const int blkGrW  = (NGRAPHS * 32 + TB - 1) / TB;
  const int blkPkW1 = (KIN * HID + TB - 1) / TB;
  const int blkPkW  = (HID * HID + TB - 1) / TB;

  // symmetric normalization coefficients
  k_deg_init <<<blkN, TB, 0, stream>>>(dinv);
  k_deg_count<<<blkE, TB, 0, stream>>>(edst, dinv);
  k_dinv     <<<blkN, TB, 0, stream>>>(dinv);

  // pre-pack all weights into fp16 WMMA B-fragment layout (tiny, stays in L2)
  k_pack_w<<<blkPkW1, TB, 0, stream>>>(w1,  w1p,  KIN);
  k_pack_w<<<blkPkW,  TB, 0, stream>>>(w2,  w2p,  HID);
  k_pack_w<<<blkPkW,  TB, 0, stream>>>(wp1, wp1p, HID);
  k_pack_w<<<blkPkW,  TB, 0, stream>>>(wp2, wp2p, HID);

  // ---- layer 1: h = x@W1 ; out = scatter(h[s]*dinv[s]*dinv[d]) + self ; BN+ReLU
  k_gemm_wmma<KIN, 0><<<blkGemm, TB, 0, stream>>>(x, w1p, nullptr, nullptr, nullptr, bufA);
  k_self_init<<<blkSelf, TB, 0, stream>>>(bufA, dinv, bufB);
  k_edge_agg <<<blkEdge, TB, 0, stream>>>(esrc, edst, bufA, dinv, bufB);
  k_post     <<<blkElem, TB, 0, stream>>>(bufB, b1, g1, be1);

  // ---- layer 2
  k_gemm_wmma<HID, 0><<<blkGemm, TB, 0, stream>>>(bufB, w2p, nullptr, nullptr, nullptr, bufA);
  k_self_init<<<blkSelf, TB, 0, stream>>>(bufA, dinv, bufB);
  k_edge_agg <<<blkEdge, TB, 0, stream>>>(esrc, edst, bufA, dinv, bufB);
  k_post     <<<blkElem, TB, 0, stream>>>(bufB, b2, g2, be2);

  // ---- projector: Linear -> BN -> ReLU -> Linear (epilogues fused into WMMA GEMM)
  k_gemm_wmma<HID, 1><<<blkGemm, TB, 0, stream>>>(bufB, wp1p, bp1, gp, bep, bufA);
  k_gemm_wmma<HID, 2><<<blkGemm, TB, 0, stream>>>(bufA, wp2p, bp2, nullptr, nullptr, bufB);

  // ---- node embeddings (row l2norm) -> d_out
  k_l2rows<<<blkRowW, TB, 0, stream>>>(bufB, emb);

  // ---- graph mean pool + l2norm
  k_pool_zero<<<blkPool, TB, 0, stream>>>(sums, cnts);
  k_pool_acc <<<blkRowW, TB, 0, stream>>>(emb, batch, sums, cnts);
  k_graph_emb<<<blkGrW, TB, 0, stream>>>(sums, cnts, gemb);
}
